// RWKV6Attention_Prototype_90314572300648
// MI455X (gfx1250) — matrix operand order repaired and checked by reading, not compiled, not run on hardware
//
#include <hip/hip_runtime.h>
#include <hip/hip_bf16.h>

// ---------------------------------------------------------------------------
// RWKV block for MI455X (gfx1250): bf16 WMMA GEMMs (async-LDS double-buffered)
// + serial WKV scan. All weights are pre-transposed to [N][K] bf16 so both
// GEMM operands stage into LDS with coalesced b128 copies (no scalar LDS
// transposes).
// ---------------------------------------------------------------------------

typedef __attribute__((ext_vector_type(16))) __bf16 vbf16x16;
typedef __attribute__((ext_vector_type(8)))  float  v8f;
typedef int v4i __attribute__((vector_size(16)));   // matches builtin param type

#define LN_EPS 1e-5f

#if __has_builtin(__builtin_amdgcn_global_load_async_to_lds_b128) && \
    __has_builtin(__builtin_amdgcn_s_wait_asynccnt)
#define HAVE_ASYNC_LDS 1
#else
#define HAVE_ASYNC_LDS 0
#endif

__device__ __forceinline__ unsigned short f2bf(float f) {
    // round-to-nearest-even fp32 -> bf16
    unsigned int u = __float_as_uint(f);
    u += 0x7FFFu + ((u >> 16) & 1u);
    return (unsigned short)(u >> 16);
}

#if HAVE_ASYNC_LDS
__device__ __forceinline__ void async_cp16(const unsigned short* g, unsigned short* l) {
    // per-lane 16B global -> LDS copy, tracked with ASYNCcnt
    __builtin_amdgcn_global_load_async_to_lds_b128(
        (__attribute__((address_space(1))) v4i*)(v4i*)(void*)g,
        (__attribute__((address_space(3))) v4i*)(v4i*)(void*)l, 0, 0);
}
#endif

// ------------------- fp32 [R,C] -> bf16 transposed [C,R] -------------------
// 32x32 tile per 256-thread block; coalesced f32 reads, packed bf16 writes.
__global__ void cvt_transpose_bf16(const float* __restrict__ in,
                                   unsigned short* __restrict__ out,
                                   int R, int Ccols) {
    __shared__ float tile[32][33];
    const int cb = blockIdx.x * 32;   // input col block
    const int rb = blockIdx.y * 32;   // input row block
    const int tr  = threadIdx.x >> 3;       // 0..31
    const int tc4 = (threadIdx.x & 7) * 4;  // 0..28
    float4 f = *(const float4*)(in + (size_t)(rb + tr) * Ccols + cb + tc4);
    tile[tr][tc4 + 0] = f.x;
    tile[tr][tc4 + 1] = f.y;
    tile[tr][tc4 + 2] = f.z;
    tile[tr][tc4 + 3] = f.w;
    __syncthreads();
    // out[n][k] = in[k][n]; out row = cb+tr, cols rb+tc4..+3
    float a = tile[tc4 + 0][tr];
    float b = tile[tc4 + 1][tr];
    float c = tile[tc4 + 2][tr];
    float d = tile[tc4 + 3][tr];
    uint2 p;
    p.x = (unsigned int)f2bf(a) | ((unsigned int)f2bf(b) << 16);
    p.y = (unsigned int)f2bf(c) | ((unsigned int)f2bf(d) << 16);
    *(uint2*)(out + (size_t)(cb + tr) * R + rb + tc4) = p;
}

// ------------------------------ LayerNorm ----------------------------------
__global__ void layernorm_bf16(const float* __restrict__ x,
                               const float* __restrict__ w,
                               const float* __restrict__ b,
                               unsigned short* __restrict__ out, int C) {
    __shared__ float red[256];
    const int tid = threadIdx.x;
    const float* xr = x + (size_t)blockIdx.x * C;
    float vals[8];
    float s = 0.f;
#pragma unroll
    for (int j = 0; j < 8; ++j) { vals[j] = xr[tid + j * 256]; s += vals[j]; }
    red[tid] = s;
    __syncthreads();
    for (int off = 128; off > 0; off >>= 1) {
        if (tid < off) red[tid] += red[tid + off];
        __syncthreads();
    }
    float mu = red[0] / (float)C;
    __syncthreads();
    float s2 = 0.f;
#pragma unroll
    for (int j = 0; j < 8; ++j) { float d = vals[j] - mu; s2 += d * d; }
    red[tid] = s2;
    __syncthreads();
    for (int off = 128; off > 0; off >>= 1) {
        if (tid < off) red[tid] += red[tid + off];
        __syncthreads();
    }
    float rstd = rsqrtf(red[0] / (float)C + LN_EPS);
    unsigned short* orow = out + (size_t)blockIdx.x * C;
#pragma unroll
    for (int j = 0; j < 8; ++j) {
        int idx = tid + j * 256;
        float y = (vals[j] - mu) * rstd * w[idx] + b[idx];
        orow[idx] = f2bf(y);
    }
}

// ------------------------------- WKV scan ----------------------------------
__global__ void wkv_scan(const float* __restrict__ k,
                         float* __restrict__ v, // in: v, out: wkv
                         const float* __restrict__ time_decay,
                         const float* __restrict__ time_first,
                         int Bv, int Tv, int Cv) {
    int idx = blockIdx.x * blockDim.x + threadIdx.x;
    if (idx >= Bv * Cv) return;
    int b = idx / Cv, hs = idx % Cv;
    float decay = __expf(-__expf(time_decay[hs]));
    float bonus = __expf(time_first[hs]);
    size_t base = (size_t)b * Tv * Cv + hs;
    float state = 0.f;
    for (int t = 0; t < Tv; ++t) {
        size_t off = base + (size_t)t * Cv;
        float vt = v[off];
        float kv = __expf(k[off]) * vt;
        state = decay * state + kv;
        float wkv = state;
        if (t == 0) wkv += bonus * vt;  // t=0 bonus added to wkv only
        v[off] = wkv;
    }
}

// --------------------------- sigmoid(r) * wkv ------------------------------
__global__ void rwkv_gate(const float* __restrict__ r,
                          const float* __restrict__ wkv,
                          unsigned short* __restrict__ out, int n) {
    int i = blockIdx.x * blockDim.x + threadIdx.x;
    if (i < n) {
        float rv = r[i];
        float sg = 1.f / (1.f + __expf(-rv));
        out[i] = f2bf(sg * wkv[i]);
    }
}

// --------------------------- bf16 WMMA GEMM --------------------------------
// D = A[M,K] * Bt[N,K]^T (+ epilogue). Block tile 128x128x64, 256 threads =
// 8 waves (wave32), wave tile 32x64 => 2x4 v_wmma_f32_16x16x32_bf16 per
// 32-wide k-step, 16 WMMAs per barrier. Double-buffered LDS, filled with
// GLOBAL_LOAD_ASYNC_TO_LDS_B128 when available.
// MODE 0: store f32   MODE 1: store Cbase + acc (f32)   MODE 2: bf16(relu^2)
#define BM 128
#define BN 128
#define BK 64

union FragAB { vbf16x16 v; uint4 q[2]; };
union FragC  { v8f v; float f[8]; };

template <int MODE>
__global__ __launch_bounds__(256)
void gemm_bf16_wmma(const unsigned short* __restrict__ A,   // [M,K] bf16
                    const unsigned short* __restrict__ Bt,  // [N,K] bf16
                    const float* __restrict__ Cbase,
                    float* __restrict__ Dout_f32,
                    unsigned short* __restrict__ Dout_bf,
                    int M, int N, int K) {
    __shared__ unsigned short lA[2][BM][BK];  // 2 x 16 KB
    __shared__ unsigned short lB[2][BN][BK];  // 2 x 16 KB

    const int tid  = threadIdx.x;
    const int wave = tid >> 5;
    const int lane = tid & 31;
    const int bm = blockIdx.x * BM;
    const int bn = blockIdx.y * BN;
    const int wm = (wave & 3) * 32;   // wave row offset inside block tile
    const int wn = (wave >> 2) * 64;  // wave col offset inside block tile

    FragC acc[2][4];
#pragma unroll
    for (int i = 0; i < 2; ++i)
#pragma unroll
        for (int j = 0; j < 4; ++j)
#pragma unroll
            for (int e = 0; e < 8; ++e) acc[i][j].f[e] = 0.f;

    // cooperative staging: thread -> (row, 8-elem column chunk), 4 row passes
    const int lrow = tid >> 3;        // 0..31
    const int lcol = (tid & 7) * 8;   // 0..56

    auto load_tile = [&](int buf, int kk) {
#pragma unroll
        for (int p = 0; p < 4; ++p) {
            const unsigned short* ga = A  + (size_t)(bm + lrow + p * 32) * K + kk + lcol;
            const unsigned short* gb = Bt + (size_t)(bn + lrow + p * 32) * K + kk + lcol;
            unsigned short* la = &lA[buf][lrow + p * 32][lcol];
            unsigned short* lb = &lB[buf][lrow + p * 32][lcol];
#if HAVE_ASYNC_LDS
            async_cp16(ga, la);
            async_cp16(gb, lb);
#else
            *(uint4*)la = *(const uint4*)ga;
            *(uint4*)lb = *(const uint4*)gb;
#endif
        }
    };

    const int l15 = lane & 15;
    const int hiA = (lane >> 4) * 8;   // A K-chunk select per lane half
    const int hiB = (lane >> 4) * 8;   // B K-chunk select per lane half
    const int nk = K / BK;

    load_tile(0, 0);
#if HAVE_ASYNC_LDS
    __builtin_amdgcn_s_wait_asynccnt(0);
#endif
    __syncthreads();

    for (int it = 0; it < nk; ++it) {
        if (it + 1 < nk) load_tile((it + 1) & 1, (it + 1) * BK);  // overlap

        const unsigned short (*tA)[BK] = lA[it & 1];
        const unsigned short (*tB)[BK] = lB[it & 1];
#pragma unroll
        for (int ks = 0; ks < BK; ks += 32) {
            FragAB afr[2], bfr[4];
#pragma unroll
            for (int i = 0; i < 2; ++i) {
                const unsigned short* src = &tA[wm + i * 16 + l15][ks + hiA];
                afr[i].q[0] = *(const uint4*)(src);       // K 0..7  (or 8..15)
                afr[i].q[1] = *(const uint4*)(src + 16);  // K 16..23 (or 24..31)
            }
#pragma unroll
            for (int j = 0; j < 4; ++j) {
                const unsigned short* src = &tB[wn + j * 16 + l15][ks + hiB];
                bfr[j].q[0] = *(const uint4*)(src);       // K 0..7  (or 8..15)
                bfr[j].q[1] = *(const uint4*)(src + 16);  // K 16..23 (or 24..31)
            }
#pragma unroll
            for (int i = 0; i < 2; ++i)
#pragma unroll
                for (int j = 0; j < 4; ++j)
                    acc[i][j].v = __builtin_amdgcn_wmma_f32_16x16x32_bf16(
                        false, afr[i].v, false, bfr[j].v,
                        (short)0, acc[i][j].v, false, false);
        }
#if HAVE_ASYNC_LDS
        __builtin_amdgcn_s_wait_asynccnt(0);
#endif
        __syncthreads();
    }

    // ---- epilogue: C/D layout — VGPR e holds M=e (lanes 0-15) / M=8+e ----
    const int hiC = (lane >> 4) * 8;
#pragma unroll
    for (int i = 0; i < 2; ++i) {
#pragma unroll
        for (int j = 0; j < 4; ++j) {
            int row0 = bm + wm + i * 16 + hiC;
            int col  = bn + wn + j * 16 + l15;
#pragma unroll
            for (int e = 0; e < 8; ++e) {
                size_t off = (size_t)(row0 + e) * N + col;
                float val = acc[i][j].f[e];
                if (MODE == 0) {
                    Dout_f32[off] = val;
                } else if (MODE == 1) {
                    Dout_f32[off] = Cbase[off] + val;
                } else {
                    float rv = val > 0.f ? val : 0.f;
                    Dout_bf[off] = f2bf(rv * rv);
                }
            }
        }
    }
}

// ---------------------------------------------------------------------------
extern "C" void kernel_launch(void* const* d_in, const int* in_sizes, int n_in,
                              void* d_out, int out_size, void* d_ws, size_t ws_size,
                              hipStream_t stream) {
    const float* x    = (const float*)d_in[0];
    const float* td   = (const float*)d_in[1];
    const float* tf   = (const float*)d_in[2];
    const float* Wk   = (const float*)d_in[3];
    const float* Wv   = (const float*)d_in[4];
    const float* Wr   = (const float*)d_in[5];
    const float* Wo   = (const float*)d_in[6];
    const float* Wfk  = (const float*)d_in[7];
    const float* Wfv  = (const float*)d_in[8];
    const float* ln1w = (const float*)d_in[9];
    const float* ln1b = (const float*)d_in[10];
    const float* ln2w = (const float*)d_in[11];
    const float* ln2b = (const float*)d_in[12];
    float* out = (float*)d_out;

    const int Bs = 4, Ts = 2048, Cs = 2048;
    const int Mrows = Bs * Ts;   // 8192
    const int F = 4 * Cs;        // 8192

    // ---- workspace layout (~320 MB, aggressive reuse) ----
    char* ws = (char*)d_ws;
    const size_t MB = 1024ull * 1024ull;
    unsigned short* wk_bf  = (unsigned short*)(ws + 0   * MB);   //  8 MB  Wk^T
    unsigned short* wv_bf  = (unsigned short*)(ws + 8   * MB);   //  8 MB  Wv^T
    unsigned short* wr_bf  = (unsigned short*)(ws + 16  * MB);   //  8 MB  Wr^T
    unsigned short* wo_bf  = (unsigned short*)(ws + 24  * MB);   //  8 MB  Wo^T
    unsigned short* wfk_bf = (unsigned short*)(ws + 32  * MB);   // 32 MB  Wfk^T
    unsigned short* wfv_bf = (unsigned short*)(ws + 64  * MB);   // 32 MB  Wfv^T
    unsigned short* xl_bf  = (unsigned short*)(ws + 96  * MB);   // 32 MB (later: rwkv bf16)
    float*          kbuf   = (float*)         (ws + 128 * MB);   // 64 MB (later: xl2 bf16)
    float*          vbuf   = (float*)         (ws + 192 * MB);   // 64 MB (v, then wkv in place)
    float*          rbuf   = (float*)         (ws + 256 * MB);   // 64 MB
    unsigned short* xl2_bf = (unsigned short*)(ws + 128 * MB);   // reuses k region
    unsigned short* h_bf   = (unsigned short*)(ws + 192 * MB);   // 128 MB, reuses v+r regions

    dim3 blk(256);
    auto cvtT = [&](const float* src, unsigned short* dst, int R, int Ccols) {
        // src: [R,Ccols] f32 -> dst: [Ccols,R] bf16
        cvt_transpose_bf16<<<dim3(Ccols / 32, R / 32), blk, 0, stream>>>(src, dst, R, Ccols);
    };

    // 1) weights -> transposed bf16 (one-off, bandwidth bound ~15 us)
    cvtT(Wk,  wk_bf,  Cs, Cs);
    cvtT(Wv,  wv_bf,  Cs, Cs);
    cvtT(Wr,  wr_bf,  Cs, Cs);
    cvtT(Wo,  wo_bf,  Cs, Cs);
    cvtT(Wfk, wfk_bf, Cs, F);   // [C,4C] -> [4C,C]
    cvtT(Wfv, wfv_bf, F, Cs);   // [4C,C] -> [C,4C]

    // 2) LN1 -> xl (bf16)
    layernorm_bf16<<<dim3(Mrows), blk, 0, stream>>>(x, ln1w, ln1b, xl_bf, Cs);

    // 3) k, v, r = xl @ {Wk, Wv, Wr}   (8192 x 2048 x 2048, bf16 WMMA)
    dim3 gC(Mrows / BM, Cs / BN);
    gemm_bf16_wmma<0><<<gC, blk, 0, stream>>>(xl_bf, wk_bf, nullptr, kbuf, nullptr, Mrows, Cs, Cs);
    gemm_bf16_wmma<0><<<gC, blk, 0, stream>>>(xl_bf, wv_bf, nullptr, vbuf, nullptr, Mrows, Cs, Cs);
    gemm_bf16_wmma<0><<<gC, blk, 0, stream>>>(xl_bf, wr_bf, nullptr, rbuf, nullptr, Mrows, Cs, Cs);

    // 4) serial WKV scan; wkv overwrites v in place
    wkv_scan<<<dim3((Bs * Cs) / 256), blk, 0, stream>>>(kbuf, vbuf, td, tf, Bs, Ts, Cs);

    // 5) rwkv = sigmoid(r) * wkv -> bf16 (reuse xl_bf storage)
    rwkv_gate<<<dim3((Mrows * Cs) / 256), blk, 0, stream>>>(rbuf, vbuf, xl_bf, Mrows * Cs);

    // 6) x1 = x + rwkv @ Wo  -> d_out
    gemm_bf16_wmma<1><<<gC, blk, 0, stream>>>(xl_bf, wo_bf, x, out, nullptr, Mrows, Cs, Cs);

    // 7) LN2 on x1 -> xl2 (bf16, reuses k region)
    layernorm_bf16<<<dim3(Mrows), blk, 0, stream>>>(out, ln2w, ln2b, xl2_bf, Cs);

    // 8) h = relu(xl2 @ Wfk)^2 -> bf16   (8192 x 8192 x 2048)
    dim3 gF1(Mrows / BM, F / BN);
    gemm_bf16_wmma<2><<<gF1, blk, 0, stream>>>(xl2_bf, wfk_bf, nullptr, nullptr, h_bf, Mrows, F, Cs);

    // 9) out = x1 + h @ Wfv              (8192 x 2048 x 8192, in-place accumulate)
    gemm_bf16_wmma<1><<<gC, blk, 0, stream>>>(h_bf, wfv_bf, out, out, nullptr, Mrows, Cs, F);
}
